// SlotAttention_23321672417764
// MI455X (gfx1250) — matrix operand order repaired and checked by reading, not compiled, not run on hardware
//
#include <hip/hip_runtime.h>

typedef __attribute__((ext_vector_type(16))) _Float16 v16h;
typedef __attribute__((ext_vector_type(8)))  _Float16 v8h;
typedef __attribute__((ext_vector_type(8)))  float    v8f;

#define B_ 32
#define N_ 4096
#define K_ 8
#define D_ 256
#define H_ 512

// ---------------- WMMA helpers (CDNA5 wave32, 16x16x32 f16 -> f32) -------

__device__ __forceinline__ v8f wmma_f16(v16h a, v16h b, v8f c) {
  return __builtin_amdgcn_wmma_f32_16x16x32_f16(
      /*neg_a=*/false, a, /*neg_b=*/false, b,
      /*c_mod=*/(short)0, c, /*reuse_a=*/false, /*reuse_b=*/false);
}

// A fragment: lane holds row m = lane&15, 16 contiguous K values at
// ks + 16*(lane>>4), where kb = ks + 8*(lane>>4).
__device__ __forceinline__ v16h afrag(const _Float16* A, int lda, int m, int kb) {
  const _Float16* p = A + m * lda + kb + (kb & 8);
  v16h r;
#pragma unroll
  for (int i = 0; i < 16; ++i) { r[i] = p[i]; }
  return r;
}

// B fragment: lane holds col n = lane&15, 16 contiguous K values at
// ks + 16*(lane>>4).
__device__ __forceinline__ v16h bfrag(const _Float16* p) {
  return *(const v16h*)p;
}

__device__ __forceinline__ float wsum32(float v) {
#pragma unroll
  for (int m = 16; m >= 1; m >>= 1) v += __shfl_xor(v, m, 32);
  return v;
}

#define LDA0 260   // 16-row f16 A buffers, padded stride

// LN of 8 slot rows (wave w -> row w) from LDS f32 into A0 f16 rows 0-7.
__device__ __forceinline__ void ln_rows8(const float* slots_s, _Float16* A0,
                                         const float* g, const float* bb,
                                         int lane, int wv) {
  int row = wv;
  float xv[8];
#pragma unroll
  for (int i = 0; i < 8; ++i) xv[i] = slots_s[row * D_ + lane * 8 + i];
  float s = 0.f, sq = 0.f;
#pragma unroll
  for (int i = 0; i < 8; ++i) { s += xv[i]; sq += xv[i] * xv[i]; }
  s = wsum32(s); sq = wsum32(sq);
  float mean = s * (1.f / 256.f);
  float var  = sq * (1.f / 256.f) - mean * mean;
  float rs   = rsqrtf(var + 1e-5f);
#pragma unroll
  for (int i = 0; i < 8; ++i) {
    int j = lane * 8 + i;
    A0[row * LDA0 + j] = (_Float16)((xv[i] - mean) * rs * g[j] + bb[j]);
  }
}

// q = A0 @ Wq^T + bq, scaled by 1/sqrt(D); stored f16 to global [16][256].
__device__ __forceinline__ void qproj(const _Float16* A0, const _Float16* Wq16,
                                      const float* bq, _Float16* qg_b,
                                      int lane, int wv) {
  const int m = lane & 15, half = lane >> 4;
  for (int t = wv; t < 16; t += 8) {
    int j = t * 16 + m;
    const _Float16* Brow = Wq16 + (size_t)j * D_;
    v8f acc = {};
#pragma unroll
    for (int ks = 0; ks < 256; ks += 32)
      acc = wmma_f16(afrag(A0, LDA0, m, ks + half * 8), bfrag(Brow + ks + half * 16), acc);
    float bj = bq[j];
#pragma unroll
    for (int r = 0; r < 8; ++r)
      qg_b[(r + half * 8) * D_ + j] = (_Float16)((acc[r] + bj) * 0.0625f);
  }
}

// ---------------- weight f32 -> f16 conversion ---------------------------

__global__ void cvt_f32_f16(const float* __restrict__ s, _Float16* __restrict__ d, int n) {
  int i = blockIdx.x * blockDim.x + threadIdx.x;
  if (i < n) d[i] = (_Float16)s[i];
}

// ---------------- fused LayerNorm + k/v projection ------------------------
// Block: 256 thr (8 waves), 64 rows of x. Outputs k row-major [B*N][D] f16
// and v transposed [B][D][N] f16.

#define KV_ROWS 64
#define KV_LDA  264

__global__ __launch_bounds__(256) void kv_kernel(
    const float* __restrict__ x,
    const float* __restrict__ ln_g, const float* __restrict__ ln_b,
    const _Float16* __restrict__ Wk16, const float* __restrict__ bk,
    const _Float16* __restrict__ Wv16, const float* __restrict__ bv,
    _Float16* __restrict__ kout, _Float16* __restrict__ vT) {
  __shared__ _Float16 a_sh[KV_ROWS * KV_LDA];

  const int lane = threadIdx.x & 31;
  const int wv   = threadIdx.x >> 5;
  const int m    = lane & 15;
  const int half = lane >> 4;
  const int grow0 = blockIdx.x * KV_ROWS;

  float gv[8], bv8[8];
#pragma unroll
  for (int i = 0; i < 8; ++i) { gv[i] = ln_g[lane * 8 + i]; bv8[i] = ln_b[lane * 8 + i]; }

  for (int rr = 0; rr < 8; ++rr) {
    int row = wv * 8 + rr;
    const float* xp = x + (size_t)(grow0 + row) * D_ + lane * 8;
    float xv[8];
#pragma unroll
    for (int i = 0; i < 8; ++i) xv[i] = xp[i];
    float s = 0.f, sq = 0.f;
#pragma unroll
    for (int i = 0; i < 8; ++i) { s += xv[i]; sq += xv[i] * xv[i]; }
    s = wsum32(s); sq = wsum32(sq);
    float mean = s * (1.f / 256.f);
    float var  = sq * (1.f / 256.f) - mean * mean;
    float rs   = rsqrtf(var + 1e-5f);
#pragma unroll
    for (int i = 0; i < 8; ++i)
      a_sh[row * KV_LDA + lane * 8 + i] = (_Float16)((xv[i] - mean) * rs * gv[i] + bv8[i]);
  }
  __syncthreads();

  for (int t = wv; t < 128; t += 8) {
    int mt = t >> 5;
    int o  = t & 31;
    const _Float16* W   = (o < 16) ? Wk16 : Wv16;
    const float*    bia = (o < 16) ? bk : bv;
    int j = (o & 15) * 16 + m;
    const _Float16* Brow = W + (size_t)j * D_;
    v8f acc = {};
#pragma unroll
    for (int ks = 0; ks < 256; ks += 32) {
      v16h a = afrag(a_sh + (mt * 16) * KV_LDA, KV_LDA, m, ks + half * 8);
      v16h b = bfrag(Brow + ks + half * 16);
      acc = wmma_f16(a, b, acc);
    }
    float bb = bia[j];
    if (o < 16) {
#pragma unroll
      for (int r = 0; r < 8; ++r) {
        int grow = grow0 + mt * 16 + r + half * 8;
        kout[(size_t)grow * D_ + j] = (_Float16)(acc[r] + bb);
      }
    } else {
      int b_ = grow0 / N_;
      int nl = (grow0 % N_) + mt * 16 + half * 8;
      v8h pk;
#pragma unroll
      for (int r = 0; r < 8; ++r) pk[r] = (_Float16)(acc[r] + bb);
      *(v8h*)(vT + (size_t)b_ * D_ * N_ + (size_t)j * N_ + nl) = pk;
    }
  }
}

// ---------------- prep: slots init + LN + q projection --------------------

__global__ __launch_bounds__(256) void prep_kernel(
    const float* __restrict__ slots_init,
    const float* __restrict__ ln_s_g, const float* __restrict__ ln_s_b,
    const _Float16* __restrict__ Wq16, const float* __restrict__ bq,
    float* __restrict__ slots_ws, float* __restrict__ upd_ws,
    _Float16* __restrict__ qg) {
  __shared__ float    slots[K_ * D_];
  __shared__ _Float16 A0[16 * LDA0];
  const int tid = threadIdx.x, lane = tid & 31, wv = tid >> 5;
  const int b_ = blockIdx.x;

  for (int i = tid; i < K_ * D_; i += 256) {
    float v = slots_init[(size_t)b_ * K_ * D_ + i];
    slots[i] = v;
    slots_ws[(size_t)b_ * K_ * D_ + i] = v;
    upd_ws[(size_t)b_ * K_ * D_ + i] = 0.f;
  }
  for (int i = tid; i < 8 * 256; i += 256)
    A0[(8 + (i >> 8)) * LDA0 + (i & 255)] = (_Float16)0.f;
  __syncthreads();
  ln_rows8(slots, A0, ln_s_g, ln_s_b, lane, wv);
  __syncthreads();
  qproj(A0, Wq16, bq, qg + (size_t)b_ * 16 * D_, lane, wv);
}

// ---------------- attention: 8 blocks per batch, 512 keys each ------------

__global__ __launch_bounds__(256) void attn_kernel(
    const _Float16* __restrict__ qg,
    const _Float16* __restrict__ kbuf, const _Float16* __restrict__ vTbuf,
    float* __restrict__ upd_ws) {
  __shared__ _Float16 qs[16 * LDA0];       // q for this batch
  __shared__ _Float16 atts[8 * 16 * 32];   // per-wave attn scratch
  __shared__ float    upd_s[K_ * D_];      // block-local reduction

  const int tid  = threadIdx.x;
  const int lane = tid & 31;
  const int wv   = tid >> 5;
  const int m    = lane & 15;
  const int half = lane >> 4;
  const int b_   = blockIdx.x >> 3;
  const int ch   = blockIdx.x & 7;
  const _Float16* kb_ = kbuf  + (size_t)b_ * N_ * D_;
  const _Float16* vTb = vTbuf + (size_t)b_ * D_ * N_;
  const _Float16* qgb = qg    + (size_t)b_ * 16 * D_;

  for (int i = tid; i < 16 * 256; i += 256)
    qs[(i >> 8) * LDA0 + (i & 255)] = qgb[i];
  for (int i = tid; i < K_ * D_; i += 256) upd_s[i] = 0.f;
  __syncthreads();

  _Float16* att = atts + wv * (16 * 32);
  v8f accU[16] = {};
  int n0s = ch * 512 + wv * 64;
  for (int n0 = n0s; n0 < n0s + 64; n0 += 32) {
#pragma unroll
    for (int h2 = 0; h2 < 2; ++h2) {
      int key = n0 + h2 * 16 + m;
      const _Float16* Brow = kb_ + (size_t)key * D_;
      v8f accL = {};
#pragma unroll
      for (int ks = 0; ks < 256; ks += 32)
        accL = wmma_f16(afrag(qs, LDA0, m, ks + half * 8), bfrag(Brow + ks + half * 16), accL);
      // softmax across slots (rows 0-7): entirely within lanes 0-15
      if (half == 0) {
        float mx = accL[0];
#pragma unroll
        for (int r = 1; r < 8; ++r) mx = fmaxf(mx, accL[r]);
        float e[8], ssum = 0.f;
#pragma unroll
        for (int r = 0; r < 8; ++r) { e[r] = __expf(accL[r] - mx); ssum += e[r]; }
        float inv = 1.f / ssum;
#pragma unroll
        for (int r = 0; r < 8; ++r) att[r * 32 + h2 * 16 + m] = (_Float16)(e[r] * inv);
      } else {
#pragma unroll
        for (int r = 0; r < 8; ++r) att[(r + 8) * 32 + h2 * 16 + m] = (_Float16)0.f;
      }
    }
    asm volatile("s_wait_dscnt 0x0" ::: "memory");   // wave-local LDS transpose bounce
    v16h aA = afrag(att, 32, m, half * 8);
#pragma unroll
    for (int t = 0; t < 16; ++t) {
      const _Float16* Brow = vTb + (size_t)(t * 16 + m) * N_ + n0 + half * 16;
      accU[t] = wmma_f16(aA, bfrag(Brow), accU[t]);
    }
  }
  // block-level reduction via LDS float atomics (rows 0-7 in lanes 0-15)
  if (half == 0) {
#pragma unroll
    for (int t = 0; t < 16; ++t)
#pragma unroll
      for (int r = 0; r < 8; ++r)
        atomicAdd(&upd_s[r * D_ + t * 16 + m], accU[t][r]);
  }
  __syncthreads();
  // cross-block accumulation into global upd
  for (int i = tid; i < K_ * D_; i += 256)
    atomicAdd(&upd_ws[(size_t)b_ * K_ * D_ + i], upd_s[i]);
}

// ---------------- GRU + MLP update (+ next-iter LN/q) ---------------------

__global__ __launch_bounds__(256) void update_kernel(
    const float* __restrict__ ln_s_g,  const float* __restrict__ ln_s_b,
    const float* __restrict__ ln_ff_g, const float* __restrict__ ln_ff_b,
    const _Float16* __restrict__ Wq16,  const float* __restrict__ bq,
    const _Float16* __restrict__ Wih16, const float* __restrict__ b_ih,
    const _Float16* __restrict__ Whh16, const float* __restrict__ b_hh,
    const _Float16* __restrict__ W116,  const float* __restrict__ b1,
    const _Float16* __restrict__ W216,  const float* __restrict__ b2,
    float* __restrict__ slots_ws, float* __restrict__ upd_ws,
    _Float16* __restrict__ qg, int is_last, float* __restrict__ out) {
#define LDH 528
  __shared__ float    slots[K_ * D_];
  __shared__ _Float16 A0[16 * LDA0];
  __shared__ _Float16 A1[16 * LDA0];
  __shared__ _Float16 A2[16 * LDA0];
  __shared__ _Float16 Ah[16 * LDH];

  const int tid  = threadIdx.x;
  const int lane = tid & 31;
  const int wv   = tid >> 5;
  const int m    = lane & 15;
  const int half = lane >> 4;
  const int b_   = blockIdx.x;
  float* upd_g   = upd_ws + (size_t)b_ * K_ * D_;

  for (int i = tid; i < K_ * D_; i += 256)
    slots[i] = slots_ws[(size_t)b_ * K_ * D_ + i];
  for (int i = tid; i < 8 * 256; i += 256) {
    int r = 8 + (i >> 8), c = i & 255;
    A0[r * LDA0 + c] = (_Float16)0.f;
    A1[r * LDA0 + c] = (_Float16)0.f;
    A2[r * LDA0 + c] = (_Float16)0.f;
  }
  __syncthreads();

  // upd16 / slots16 f16 A matrices
  {
    int row = tid >> 5;
#pragma unroll
    for (int i = 0; i < 8; ++i) {
      int j = lane * 8 + i;
      A1[row * LDA0 + j] = (_Float16)upd_g[row * D_ + j];
      A2[row * LDA0 + j] = (_Float16)slots[row * D_ + j];
    }
  }
  __syncthreads();

  // GRU cell
  for (int t = wv; t < 16; t += 8) {
    int j = t * 16 + m;
    const _Float16* Bir = Wih16 + (size_t)(j)       * D_;
    const _Float16* Biz = Wih16 + (size_t)(256 + j) * D_;
    const _Float16* Bin = Wih16 + (size_t)(512 + j) * D_;
    const _Float16* Bhr = Whh16 + (size_t)(j)       * D_;
    const _Float16* Bhz = Whh16 + (size_t)(256 + j) * D_;
    const _Float16* Bhn = Whh16 + (size_t)(512 + j) * D_;
    v8f ir = {}, iz = {}, in_ = {}, hr = {}, hz = {}, hn = {};
#pragma unroll
    for (int ks = 0; ks < 256; ks += 32) {
      v16h a1 = afrag(A1, LDA0, m, ks + half * 8);
      v16h a2 = afrag(A2, LDA0, m, ks + half * 8);
      int off = ks + half * 16;
      ir  = wmma_f16(a1, bfrag(Bir + off), ir);
      iz  = wmma_f16(a1, bfrag(Biz + off), iz);
      in_ = wmma_f16(a1, bfrag(Bin + off), in_);
      hr  = wmma_f16(a2, bfrag(Bhr + off), hr);
      hz  = wmma_f16(a2, bfrag(Bhz + off), hz);
      hn  = wmma_f16(a2, bfrag(Bhn + off), hn);
    }
    float bir = b_ih[j], biz = b_ih[256 + j], bin = b_ih[512 + j];
    float bhr = b_hh[j], bhz = b_hh[256 + j], bhn = b_hh[512 + j];
    if (half == 0) {
#pragma unroll
      for (int r = 0; r < 8; ++r) {
        float rg = 1.f / (1.f + __expf(-(ir[r] + bir + hr[r] + bhr)));
        float zg = 1.f / (1.f + __expf(-(iz[r] + biz + hz[r] + bhz)));
        float ng = tanhf(in_[r] + bin + rg * (hn[r] + bhn));
        float h0 = slots[r * D_ + j];
        slots[r * D_ + j] = (1.f - zg) * ng + zg * h0;
      }
    }
  }
  __syncthreads();

  // LN_ff -> A0
  ln_rows8(slots, A0, ln_ff_g, ln_ff_b, lane, wv);
  __syncthreads();

  // hdd = relu(A0 @ W1^T + b1) -> Ah
  for (int t = wv; t < 32; t += 8) {
    int j = t * 16 + m;
    const _Float16* Brow = W116 + (size_t)j * D_;
    v8f acc = {};
#pragma unroll
    for (int ks = 0; ks < 256; ks += 32)
      acc = wmma_f16(afrag(A0, LDA0, m, ks + half * 8), bfrag(Brow + ks + half * 16), acc);
    float bj = b1[j];
#pragma unroll
    for (int r = 0; r < 8; ++r) {
      float hd = acc[r] + bj;
      Ah[(r + half * 8) * LDH + j] = (_Float16)(hd > 0.f ? hd : 0.f);
    }
  }
  __syncthreads();

  // slots += Ah @ W2^T + b2
  for (int t = wv; t < 16; t += 8) {
    int j = t * 16 + m;
    const _Float16* Brow = W216 + (size_t)j * H_;
    v8f acc = {};
#pragma unroll
    for (int ks = 0; ks < 512; ks += 32)
      acc = wmma_f16(afrag(Ah, LDH, m, ks + half * 8), bfrag(Brow + ks + half * 16), acc);
    float bj = b2[j];
    if (half == 0) {
#pragma unroll
      for (int r = 0; r < 8; ++r) slots[r * D_ + j] += acc[r] + bj;
    }
  }
  __syncthreads();

  // write back slots; set up next iteration (or final output)
  for (int i = tid; i < K_ * D_; i += 256) {
    float v = slots[i];
    slots_ws[(size_t)b_ * K_ * D_ + i] = v;
    if (is_last) out[(size_t)b_ * K_ * D_ + i] = v;
    else upd_g[i] = 0.f;
  }
  if (!is_last) {
    ln_rows8(slots, A0, ln_s_g, ln_s_b, lane, wv);
    __syncthreads();
    qproj(A0, Wq16, bq, qg + (size_t)b_ * 16 * D_, lane, wv);
  }
#undef LDH
}

// ---------------- host launcher -------------------------------------------

extern "C" void kernel_launch(void* const* d_in, const int* in_sizes, int n_in,
                              void* d_out, int out_size, void* d_ws, size_t ws_size,
                              hipStream_t stream) {
  (void)in_sizes; (void)n_in; (void)out_size; (void)ws_size;
  const float* x          = (const float*)d_in[0];
  const float* slots_init = (const float*)d_in[1];
  const float* ln_in_g    = (const float*)d_in[2];
  const float* ln_in_b    = (const float*)d_in[3];
  const float* ln_s_g     = (const float*)d_in[4];
  const float* ln_s_b     = (const float*)d_in[5];
  const float* ln_ff_g    = (const float*)d_in[6];
  const float* ln_ff_b    = (const float*)d_in[7];
  const float* Wk  = (const float*)d_in[8];   const float* bk   = (const float*)d_in[9];
  const float* Wq  = (const float*)d_in[10];  const float* bq   = (const float*)d_in[11];
  const float* Wv  = (const float*)d_in[12];  const float* bv   = (const float*)d_in[13];
  const float* Wih = (const float*)d_in[14];  const float* b_ih = (const float*)d_in[15];
  const float* Whh = (const float*)d_in[16];  const float* b_hh = (const float*)d_in[17];
  const float* W1  = (const float*)d_in[18];  const float* b1   = (const float*)d_in[19];
  const float* W2  = (const float*)d_in[20];  const float* b2   = (const float*)d_in[21];

  char* ws = (char*)d_ws;
  size_t off = 0;
  auto carve = [&](size_t bytes) -> void* {
    void* p = ws + off;
    off += (bytes + 255) & ~(size_t)255;
    return p;
  };
  _Float16* k16   = (_Float16*)carve((size_t)B_ * N_ * D_ * 2);   // 64 MB
  _Float16* vT16  = (_Float16*)carve((size_t)B_ * D_ * N_ * 2);   // 64 MB
  _Float16* Wk16  = (_Float16*)carve((size_t)D_ * D_ * 2);
  _Float16* Wv16  = (_Float16*)carve((size_t)D_ * D_ * 2);
  _Float16* Wq16  = (_Float16*)carve((size_t)D_ * D_ * 2);
  _Float16* Wih16 = (_Float16*)carve((size_t)3 * D_ * D_ * 2);
  _Float16* Whh16 = (_Float16*)carve((size_t)3 * D_ * D_ * 2);
  _Float16* W116  = (_Float16*)carve((size_t)H_ * D_ * 2);
  _Float16* W216  = (_Float16*)carve((size_t)D_ * H_ * 2);
  float*    slots_ws = (float*)carve((size_t)B_ * K_ * D_ * 4);
  float*    upd_ws   = (float*)carve((size_t)B_ * K_ * D_ * 4);
  _Float16* qg       = (_Float16*)carve((size_t)B_ * 16 * D_ * 2);

  auto cv = [&](const float* s, _Float16* d, int n) {
    cvt_f32_f16<<<(n + 255) / 256, 256, 0, stream>>>(s, d, n);
  };
  cv(Wk,  Wk16,  D_ * D_);
  cv(Wv,  Wv16,  D_ * D_);
  cv(Wq,  Wq16,  D_ * D_);
  cv(Wih, Wih16, 3 * D_ * D_);
  cv(Whh, Whh16, 3 * D_ * D_);
  cv(W1,  W116,  H_ * D_);
  cv(W2,  W216,  D_ * H_);

  kv_kernel<<<(B_ * N_) / KV_ROWS, 256, 0, stream>>>(
      x, ln_in_g, ln_in_b, Wk16, bk, Wv16, bv, k16, vT16);

  prep_kernel<<<B_, 256, 0, stream>>>(
      slots_init, ln_s_g, ln_s_b, Wq16, bq, slots_ws, upd_ws, qg);

  for (int it = 0; it < 3; ++it) {
    attn_kernel<<<B_ * 8, 256, 0, stream>>>(qg, k16, vT16, upd_ws);
    update_kernel<<<B_, 256, 0, stream>>>(
        ln_s_g, ln_s_b, ln_ff_g, ln_ff_b,
        Wq16, bq, Wih16, b_ih, Whh16, b_hh, W116, b1, W216, b2,
        slots_ws, upd_ws, qg, (it == 2) ? 1 : 0, (float*)d_out);
  }
}
